// Local_Spatial_Multihead_Self_Attention_14190571946267
// MI455X (gfx1250) — matrix operand order repaired and checked by reading, not compiled, not run on hardware
//
#include <hip/hip_runtime.h>
#include <hip/hip_bf16.h>

#define CDIM   256
#define HEADS  8
#define GROUPS 32
#define WSZ    8
#define HW     65536      // 256*256
#define SEQ    64         // WSZ*WSZ tokens per window
#define DHEAD  32
#define NWIN   4096       // 4 * 32 * 32

typedef __attribute__((ext_vector_type(16))) _Float16 v16h;
typedef __attribute__((ext_vector_type(8)))  _Float16 v8h;
typedef __attribute__((ext_vector_type(8)))  float    v8f;

// LDS row strides (in f16 elements); every row start and fragment sub-offset
// is 16-byte aligned, padding avoids systematic bank conflicts.
#define XSTR 264   // 64 x 256 token-major buffers (xin / Q / K / attn-out)
#define VSTR 72    // 256 x 64 V-transposed (d-major)
#define PSTR 72    // 16 x 64 per-wave softmax P scratch

// ---------------------------------------------------------------------------
// Fragment loaders for v_wmma_f32_16x16x32_f16 (wave32).
// A (16x32, row-per-lane): lanes 0-15 hold K 0..7 & 16..23, lanes 16-31 hold
// K 8..15 & 24..31 (ISA 7.12.2 16-bit A table). Two 16B loads per lane.
__device__ __forceinline__ v16h loadA(const _Float16* row, int kh, int kbase) {
  const v8h lo = *(const v8h*)(row + kbase + kh * 8);
  const v8h hi = *(const v8h*)(row + kbase + 16 + kh * 8);
  v16h r;
#pragma unroll
  for (int j = 0; j < 8; ++j) { r[j] = lo[j]; r[8 + j] = hi[j]; }
  return r;
}
// B (32x16, col-per-lane): lanes 0-15 hold K 0..7 & 16..23, lanes 16-31 hold K 8..15 & 24..31.
__device__ __forceinline__ v16h loadB(const _Float16* col, int kh, int kbase) {
  const v8h lo = *(const v8h*)(col + kbase + kh * 8);
  const v8h hi = *(const v8h*)(col + kbase + 16 + kh * 8);
  v16h r;
#pragma unroll
  for (int j = 0; j < 8; ++j) { r[j] = lo[j]; r[8 + j] = hi[j]; }
  return r;
}

__device__ __forceinline__ v8f wmma_f16(v16h a, v16h b, v8f c) {
  return __builtin_amdgcn_wmma_f32_16x16x32_f16(false, a, false, b,
                                                (short)0, c, false, false);
}

// One projection: dst(64x256) = src(64x256) @ W^T, 64 16x16 tiles over 8
// waves, branch-free (TR selects transposed d-major store for V).
template <bool TR>
__device__ __forceinline__ void proj_tiles(const _Float16* __restrict__ src,
                                           const _Float16* __restrict__ W,
                                           _Float16* __restrict__ dst,
                                           const int dstride,
                                           const int wave, const int l16,
                                           const int kh) {
  for (int tile = wave; tile < 64; tile += 8) {
    const int mt = tile >> 4, nt = tile & 15;
    const _Float16* arow = src + (mt * 16 + l16) * XSTR;
    const _Float16* bcol = W + (nt * 16 + l16) * CDIM;   // W row n = B column n
    v8f acc = {};
#pragma unroll
    for (int kt = 0; kt < 8; ++kt)
      acc = wmma_f16(loadA(arow, kh, kt * 32), loadB(bcol, kh, kt * 32), acc);
    const int col   = nt * 16 + l16;
    const int rbase = mt * 16 + (kh ? 8 : 0);
    if (TR) {                       // V: d-major, 8 contiguous f16 -> b128
      v8h pv;
#pragma unroll
      for (int rr = 0; rr < 8; ++rr) pv[rr] = (_Float16)acc[rr];
      *(v8h*)(dst + col * dstride + rbase) = pv;
    } else {                        // Q/K: token-major (strided b16 stores)
#pragma unroll
      for (int rr = 0; rr < 8; ++rr)
        dst[(rbase + rr) * dstride + col] = (_Float16)acc[rr];
    }
  }
}

// ---------------------------------------------------------------------------
// Kernel A: GroupNorm statistics. One block per (batch, group); a group is
// 8 contiguous channels => one contiguous 2 MB span. float4 loads.
__global__ void gn_stats_kernel(const float* __restrict__ x,
                                float* __restrict__ stats) {
  const int bg = blockIdx.x;                      // b*32 + g, 0..127
  const float4* p = (const float4*)(x + (size_t)bg * 8 * HW);
  float s = 0.f, s2 = 0.f;
  const int n4 = (8 * HW) / 4;
  for (int i = threadIdx.x; i < n4; i += blockDim.x) {
    const float4 v = p[i];
    s  += v.x + v.y + v.z + v.w;
    s2 += v.x * v.x + v.y * v.y + v.z * v.z + v.w * v.w;
  }
  __shared__ float red[512];
  red[threadIdx.x] = s;
  red[256 + threadIdx.x] = s2;
  __syncthreads();
  for (int off = 128; off > 0; off >>= 1) {
    if (threadIdx.x < off) {
      red[threadIdx.x] += red[threadIdx.x + off];
      red[256 + threadIdx.x] += red[256 + threadIdx.x + off];
    }
    __syncthreads();
  }
  if (threadIdx.x == 0) {
    const float inv_n = 1.0f / (float)(8 * HW);
    const float mean  = red[0] * inv_n;
    const float var   = red[256] * inv_n - mean * mean;
    stats[bg * 2 + 0] = mean;
    stats[bg * 2 + 1] = rsqrtf(var + 1e-5f);
  }
}

// Kernel B: convert projection weights to f16; attention scale is folded
// into Wq (q = (x@Wq^T)*s == x@(s*Wq)^T).
__global__ void cvt_w_kernel(const float* __restrict__ Wq, const float* __restrict__ Wk,
                             const float* __restrict__ Wv, const float* __restrict__ Wo,
                             const float* __restrict__ scale,
                             _Float16* __restrict__ q16, _Float16* __restrict__ k16,
                             _Float16* __restrict__ v16, _Float16* __restrict__ o16) {
  const int i = blockIdx.x * blockDim.x + threadIdx.x;
  if (i < CDIM * CDIM) {
    const float s = scale[0];
    q16[i] = (_Float16)(Wq[i] * s);
    k16[i] = (_Float16)Wk[i];
    v16[i] = (_Float16)Wv[i];
    o16[i] = (_Float16)Wo[i];
  }
}

// ---------------------------------------------------------------------------
// Kernel C: fused windowed attention. One 256-thread (8 wave32) block per
// 8x8 window. ~155 KB dynamic LDS (320 KB/WGP on CDNA5 -> 2 blocks/WGP).
__global__ void __launch_bounds__(256, 1)
win_attn_kernel(const float* __restrict__ x,
                const float* __restrict__ gnw, const float* __restrict__ gnb,
                const _Float16* __restrict__ Wq16, const _Float16* __restrict__ Wk16,
                const _Float16* __restrict__ Wv16, const _Float16* __restrict__ Wo16,
                const float* __restrict__ bout, const float* __restrict__ pos,
                const float* __restrict__ stats,
                float* __restrict__ out) {
  extern __shared__ _Float16 smem[];
  _Float16* Qs = smem;                    // [64][XSTR]
  _Float16* Ks = Qs + SEQ * XSTR;         // [64][XSTR]
  _Float16* Vt = Ks + SEQ * XSTR;         // [256][VSTR]  (V transposed: d-major)
  _Float16* XA = Vt + CDIM * VSTR;        // [64][XSTR]   xin, later attn-out
  _Float16* Pw = XA + SEQ * XSTR;         // [8][16][PSTR] per-wave P scratch
  float* cscale = (float*)(Pw + 8 * 16 * PSTR);   // [256] GN per-channel scale
  float* cshift = cscale + CDIM;                  // [256] GN per-channel shift

  const int tid  = threadIdx.x;
  const int wave = tid >> 5;
  const int lane = tid & 31;
  const int l16  = lane & 15;
  const int kh   = lane >> 4;

  const int win  = blockIdx.x;            // 0..4095
  const int b    = win >> 10;
  const int wrem = win & 1023;
  const int y0   = (wrem >> 5) * WSZ;
  const int x0   = (wrem & 31) * WSZ;
  const size_t bbase = (size_t)b * CDIM * HW;

  // ---- Phase 0a: per-channel GN affine (scale/shift) into LDS -----------
  {
    const int c = tid;                    // blockDim == CDIM == 256
    const int g = c >> 3;
    const float mean = stats[(b * GROUPS + g) * 2 + 0];
    const float rstd = stats[(b * GROUPS + g) * 2 + 1];
    const float s = rstd * gnw[c];
    cscale[c] = s;
    cshift[c] = gnb[c] - mean * s;
  }
  __syncthreads();

  // ---- Phase 0b: load window (float4) + GN -> xin (f16) in LDS ----------
  {
#pragma unroll
    for (int i = 0; i < 16; ++i) {
      const int gi   = i * 256 + tid;     // float4 index: 16 per channel
      const int c    = gi >> 4;
      const int f    = gi & 15;
      const int row  = f >> 1;            // window row 0..7
      const int half = f & 1;             // which float4 within the row
      const float4 v4 = *(const float4*)(x + bbase + (size_t)c * HW +
                                         (size_t)(y0 + row) * 256 + x0 + half * 4);
      const float scl = cscale[c];
      const float sh  = cshift[c];
      const int t0 = row * 8 + half * 4;
      XA[(t0 + 0) * XSTR + c] = (_Float16)(v4.x * scl + sh);
      XA[(t0 + 1) * XSTR + c] = (_Float16)(v4.y * scl + sh);
      XA[(t0 + 2) * XSTR + c] = (_Float16)(v4.z * scl + sh);
      XA[(t0 + 3) * XSTR + c] = (_Float16)(v4.w * scl + sh);
    }
  }
  __syncthreads();

  // ---- Phase 1: Q, K, V projections (3 x 64 WMMA tiles, branch-free) ----
  proj_tiles<false>(XA, Wq16, Qs, XSTR, wave, l16, kh);
  proj_tiles<false>(XA, Wk16, Ks, XSTR, wave, l16, kh);
  proj_tiles<true >(XA, Wv16, Vt, VSTR, wave, l16, kh);
  __syncthreads();

  // ---- Phase 2: per-head attention (wave w -> head w) -------------------
  {
    const int h  = wave;
    const int hc = h * DHEAD;
    _Float16* P = Pw + wave * 16 * PSTR;
    const v8f zf = {};
    for (int mt = 0; mt < 4; ++mt) {
      const int rbase = mt * 16 + (kh ? 8 : 0);
      const _Float16* qrow = Qs + (mt * 16 + l16) * XSTR + hc;
      v8f accs[4] = {zf, zf, zf, zf};
      const v16h a = loadA(qrow, kh, 0);          // K = d_head = 32: one step
#pragma unroll
      for (int nt2 = 0; nt2 < 4; ++nt2) {
        const v16h bm = loadB(Ks + (nt2 * 16 + l16) * XSTR + hc, kh, 0);
        accs[nt2] = wmma_f16(a, bm, accs[nt2]);
      }
      // positional bias (D layout: col = nt2*16+l16, row = rbase+rr)
#pragma unroll
      for (int nt2 = 0; nt2 < 4; ++nt2) {
        const int j = nt2 * 16 + l16;
#pragma unroll
        for (int rr = 0; rr < 8; ++rr)
          accs[nt2][rr] += pos[(h * SEQ + (rbase + rr)) * SEQ + j];
      }
      // row softmax: each row lives in one 16-lane half across 4 accumulators
#pragma unroll
      for (int rr = 0; rr < 8; ++rr) {
        float mx = fmaxf(fmaxf(accs[0][rr], accs[1][rr]),
                         fmaxf(accs[2][rr], accs[3][rr]));
#pragma unroll
        for (int o = 8; o > 0; o >>= 1) mx = fmaxf(mx, __shfl_xor(mx, o, 32));
        const float e0 = __expf(accs[0][rr] - mx);
        const float e1 = __expf(accs[1][rr] - mx);
        const float e2 = __expf(accs[2][rr] - mx);
        const float e3 = __expf(accs[3][rr] - mx);
        float sum = e0 + e1 + e2 + e3;
#pragma unroll
        for (int o = 8; o > 0; o >>= 1) sum += __shfl_xor(sum, o, 32);
        const float inv = 1.0f / sum;
        accs[0][rr] = e0 * inv; accs[1][rr] = e1 * inv;
        accs[2][rr] = e2 * inv; accs[3][rr] = e3 * inv;
      }
      // bounce P through per-wave LDS: D layout -> A fragment
      const int prb = kh ? 8 : 0;
#pragma unroll
      for (int nt2 = 0; nt2 < 4; ++nt2) {
        const int j = nt2 * 16 + l16;
#pragma unroll
        for (int rr = 0; rr < 8; ++rr)
          P[(prb + rr) * PSTR + j] = (_Float16)accs[nt2][rr];
      }
      asm volatile("s_wait_dscnt 0" ::: "memory");  // same-wave LDS RAW fence
      // out_tile = P(16x64) @ V_h(64x32): 2 d-tiles x 2 k-steps
#pragma unroll
      for (int dt = 0; dt < 2; ++dt) {
        v8f o = {};
#pragma unroll
        for (int kt2 = 0; kt2 < 2; ++kt2) {
          const v16h pa = loadA(P + l16 * PSTR, kh, kt2 * 32);
          const v16h vb = loadB(Vt + (hc + dt * 16 + l16) * VSTR, kh, kt2 * 32);
          o = wmma_f16(pa, vb, o);
        }
        const int ocol = hc + dt * 16 + l16;
#pragma unroll
        for (int rr = 0; rr < 8; ++rr)
          XA[(rbase + rr) * XSTR + ocol] = (_Float16)o[rr];  // attn-out (xin dead)
      }
    }
  }
  __syncthreads();

  // ---- Phase 3: output projection + bias + residual (float4 I/O) --------
  for (int tile = wave; tile < 64; tile += 8) {
    const int mt = tile >> 4;
    const int nt = tile & 15;
    const _Float16* arow = XA + (mt * 16 + l16) * XSTR;
    const _Float16* bcol = Wo16 + (nt * 16 + l16) * CDIM;
    v8f acc = {};
#pragma unroll
    for (int kt = 0; kt < 8; ++kt)
      acc = wmma_f16(loadA(arow, kh, kt * 32), loadB(bcol, kh, kt * 32), acc);
    const int c     = nt * 16 + l16;
    const float bo  = bout[c];
    const int rbase = mt * 16 + (kh ? 8 : 0);    // multiple of 8 ->
    // acc[0..7] = one contiguous window row (8 floats) at channel c
    const size_t rowidx = bbase + (size_t)c * HW +
                          (size_t)(y0 + (rbase >> 3)) * 256 + x0;
    const float4 xa = *(const float4*)(x + rowidx);
    const float4 xb = *(const float4*)(x + rowidx + 4);
    float4 oa, ob;
    oa.x = acc[0] + bo + xa.x; oa.y = acc[1] + bo + xa.y;
    oa.z = acc[2] + bo + xa.z; oa.w = acc[3] + bo + xa.w;
    ob.x = acc[4] + bo + xb.x; ob.y = acc[5] + bo + xb.y;
    ob.z = acc[6] + bo + xb.z; ob.w = acc[7] + bo + xb.w;
    *(float4*)(out + rowidx)     = oa;
    *(float4*)(out + rowidx + 4) = ob;
  }
}

// ---------------------------------------------------------------------------
extern "C" void kernel_launch(void* const* d_in, const int* in_sizes, int n_in,
                              void* d_out, int out_size, void* d_ws, size_t ws_size,
                              hipStream_t stream) {
  (void)in_sizes; (void)n_in; (void)out_size; (void)ws_size;
  const float* x    = (const float*)d_in[0];
  const float* gnw  = (const float*)d_in[1];
  const float* gnb  = (const float*)d_in[2];
  const float* Wq   = (const float*)d_in[3];
  const float* Wk   = (const float*)d_in[4];
  const float* Wv   = (const float*)d_in[5];
  const float* Wo   = (const float*)d_in[6];
  const float* bo   = (const float*)d_in[7];
  const float* pos  = (const float*)d_in[8];
  const float* scl  = (const float*)d_in[9];
  float* out = (float*)d_out;

  // workspace layout: [128*2 f32 GN stats][4 x 65536 f16 weights]
  float* stats = (float*)d_ws;
  _Float16* w16  = (_Float16*)((char*)d_ws + 1024);
  _Float16* Wq16 = w16;
  _Float16* Wk16 = w16 + CDIM * CDIM;
  _Float16* Wv16 = w16 + 2 * CDIM * CDIM;
  _Float16* Wo16 = w16 + 3 * CDIM * CDIM;

  gn_stats_kernel<<<128, 256, 0, stream>>>(x, stats);
  cvt_w_kernel<<<(CDIM * CDIM + 255) / 256, 256, 0, stream>>>(
      Wq, Wk, Wv, Wo, scl, Wq16, Wk16, Wv16, Wo16);

  const size_t smem_elems = (size_t)(3 * SEQ * XSTR) + (size_t)CDIM * VSTR +
                            (size_t)8 * 16 * PSTR;
  const size_t smem_bytes = smem_elems * sizeof(_Float16) +
                            2 * CDIM * sizeof(float);        // ~155 KB
  (void)hipFuncSetAttribute((const void*)win_attn_kernel,
                            hipFuncAttributeMaxDynamicSharedMemorySize,
                            (int)smem_bytes);
  win_attn_kernel<<<NWIN, 256, smem_bytes, stream>>>(
      x, gnw, gnb, Wq16, Wk16, Wv16, Wo16, bo, pos, stats, out);
}
